// MultiHeadAttention_2482491097474
// MI455X (gfx1250) — matrix operand order repaired and checked by reading, not compiled, not run on hardware
//
#include <hip/hip_runtime.h>
#include <hip/hip_bf16.h>
#include <stdint.h>

#define B_  4
#define S_  2048
#define D_  1024
#define H_  16
#define HS_ 64

typedef __bf16 bf16;
typedef __attribute__((ext_vector_type(16))) __bf16 v16bf;
typedef __attribute__((ext_vector_type(8)))  __bf16 v8bf;
typedef __attribute__((ext_vector_type(4)))  __bf16 v4bf;
typedef __attribute__((ext_vector_type(8)))  float  v8f;
typedef __attribute__((ext_vector_type(4)))  unsigned int v4u;
typedef __attribute__((ext_vector_type(8)))  int v8i;
typedef __attribute__((ext_vector_type(4)))  int v4i;

__device__ __forceinline__ v8f zero8() {
  v8f z;
#pragma unroll
  for (int i = 0; i < 8; ++i) z[i] = 0.0f;
  return z;
}

__device__ __forceinline__ v16bf frag_ld(const bf16* p0, const bf16* p1) {
  v8bf a = *(const v8bf*)p0;
  v8bf b = *(const v8bf*)p1;
  v16bf r;
#pragma unroll
  for (int i = 0; i < 8; ++i) { r[i] = a[i]; r[8 + i] = b[i]; }
  return r;
}

__device__ __forceinline__ v8f wmma_bf16(v16bf a, v16bf b, v8f c) {
  return __builtin_amdgcn_wmma_f32_16x16x32_bf16(false, a, false, b, (short)0, c,
                                                 false, false);
}

__device__ __forceinline__ void wave_lds_sync() {
  __builtin_amdgcn_wave_barrier();
  asm volatile("s_wait_dscnt 0" ::: "memory");
  __builtin_amdgcn_wave_barrier();
}

// ---- Tensor Data Mover: 2-D bf16 tile, global -> LDS offset 0 ----
__device__ __forceinline__ void tdm_load_2d_bf16(unsigned lds_byte_off, const void* gptr,
                                                 unsigned tile_d0, unsigned tile_d1,
                                                 unsigned tensor_d0, unsigned tensor_d1,
                                                 unsigned stride0) {
  unsigned long long ga = (unsigned long long)(uintptr_t)gptr;
  v4u g0;
  g0[0] = 1u;                                                 // count=1
  g0[1] = lds_byte_off;                                       // lds_addr
  g0[2] = (unsigned)(ga & 0xffffffffu);                       // global_addr[31:0]
  g0[3] = (unsigned)((ga >> 32) & 0x01ffffffu) | 0x80000000u; // addr[56:32] | type=2
  v8i g1;
  g1[0] = (int)(1u << 16);                                    // data_size=1 (2B)
  g1[1] = (int)((tensor_d0 & 0xffffu) << 16);
  g1[2] = (int)(((tensor_d0 >> 16) & 0xffffu) | ((tensor_d1 & 0xffffu) << 16));
  g1[3] = (int)(((tensor_d1 >> 16) & 0xffffu) | ((tile_d0 & 0xffffu) << 16));
  g1[4] = (int)(tile_d1 & 0xffffu);
  g1[5] = (int)stride0;
  g1[6] = 0;
  g1[7] = 0;
  v4i z4; z4[0] = z4[1] = z4[2] = z4[3] = 0;
#if defined(__clang_major__) && (__clang_major__ >= 23)
  v8i z8;
#pragma unroll
  for (int i = 0; i < 8; ++i) z8[i] = 0;
  __builtin_amdgcn_tensor_load_to_lds(g0, g1, z4, z4, z8, 0);
#else
  __builtin_amdgcn_tensor_load_to_lds(g0, g1, z4, z4, 0);
#endif
}

// ---------------- conversion kernels ----------------

__global__ void cvt_x_kernel(const float* __restrict__ x, bf16* __restrict__ xb, int n4) {
  int i = blockIdx.x * blockDim.x + threadIdx.x;
  if (i >= n4) return;
  float4 v = ((const float4*)x)[i];
  v4bf o;
  o[0] = (bf16)v.x; o[1] = (bf16)v.y; o[2] = (bf16)v.z; o[3] = (bf16)v.w;
  ((v4bf*)xb)[i] = o;
}

__global__ void cvt_w_kernel(const float* __restrict__ Wq, const float* __restrict__ Wk,
                             const float* __restrict__ Wv, bf16* __restrict__ wt) {
  int i = blockIdx.x * blockDim.x + threadIdx.x;
  if (i >= 3 * H_ * HS_ * D_) return;
  int d = i % D_;
  int e = (i / D_) % HS_;
  int h = (i / (D_ * HS_)) % H_;
  int m = i / (D_ * HS_ * H_);
  const float* W = (m == 0) ? Wq : ((m == 1) ? Wk : Wv);
  wt[i] = (bf16)W[(h * D_ + d) * HS_ + e];
}

__global__ void cvt_wo_kernel(const float* __restrict__ Wo, bf16* __restrict__ wot) {
  int i = blockIdx.x * blockDim.x + threadIdx.x;
  if (i >= D_ * D_) return;
  int d = i % D_;
  int e = i / D_;
  wot[i] = (bf16)Wo[d * D_ + e];
}

// ---------------- fused QKV projection ----------------
// Block (12 waves) = one (b, 16-row s-tile, 4-head group); x rows TDM-staged.
// Weight fragments register-double-buffered (prefetch distance 1 k-step).
__global__ void __launch_bounds__(384)
qkv_kernel(const bf16* __restrict__ xb, const bf16* __restrict__ wt,
           bf16* __restrict__ Q, bf16* __restrict__ K, bf16* __restrict__ Vt) {
  extern __shared__ bf16 xs[];  // [16][1024], LDS offset 0
  const int lane  = threadIdx.x & 31;
  const int l16   = lane & 15;
  const int half  = lane >> 4;
  const int klo   = half * 8;
  const int khalf = half * 16;
  const int wave  = threadIdx.x >> 5;   // 0..11
  const int m  = wave % 3;              // 0=Q 1=K 2=V
  const int hl = wave / 3;

  const int blk = blockIdx.x;           // B * S/16 * H/4 = 2048
  const int hg  = blk & 3;
  const int st  = (blk >> 2) & 127;
  const int b   = blk >> 9;
  const int sbase = st * 16;
  const int h = hg * 4 + hl;

  if (wave == 0) {
    tdm_load_2d_bf16(0u, xb + (size_t)(b * S_ + sbase) * D_,
                     D_, 16, D_, B_ * S_, D_);
    __builtin_amdgcn_s_wait_tensorcnt(0);
  }
  __syncthreads();

  v8f acc[4];
#pragma unroll
  for (int n = 0; n < 4; ++n) acc[n] = zero8();

  const bf16* arow  = xs + l16 * D_;
  const bf16* wbase = wt + (size_t)((m * H_ + h) * HS_ + l16) * D_;

  v16bf bwa[4], bwb[4];
  auto ldB = [&](int kb, v16bf* dst) {
#pragma unroll
    for (int n = 0; n < 4; ++n) {
      const bf16* wr = wbase + (size_t)(n * 16) * D_ + kb;
      dst[n] = frag_ld(wr + klo, wr + 16 + klo);
    }
  };
  auto step = [&](int kb, const v16bf* bw) {
    v16bf aX = frag_ld(arow + kb + klo, arow + kb + 16 + klo);  // LDS
#pragma unroll
    for (int n = 0; n < 4; ++n) acc[n] = wmma_bf16(aX, bw[n], acc[n]);
  };

  ldB(0, bwa);
#pragma unroll 1
  for (int kk = 0; kk < D_ / 32; kk += 2) {
    ldB((kk + 1) * 32, bwb);          // prefetch next while computing current
    step(kk * 32, bwa);
    if (kk + 2 < D_ / 32) ldB((kk + 2) * 32, bwa);
    step((kk + 1) * 32, bwb);
  }

  const size_t bh = (size_t)(b * H_ + h);
  if (m == 0) {
    // fold 1/sqrt(HS) into Q
#pragma unroll
    for (int n = 0; n < 4; ++n)
#pragma unroll
      for (int r = 0; r < 8; ++r)
        Q[(bh * S_ + sbase + r + half * 8) * HS_ + n * 16 + l16] =
            (bf16)(acc[n][r] * 0.125f);
  } else if (m == 1) {
#pragma unroll
    for (int n = 0; n < 4; ++n)
#pragma unroll
      for (int r = 0; r < 8; ++r)
        K[(bh * S_ + sbase + r + half * 8) * HS_ + n * 16 + l16] = (bf16)acc[n][r];
  } else {
#pragma unroll
    for (int n = 0; n < 4; ++n)
#pragma unroll
      for (int r = 0; r < 8; ++r)
        Vt[(bh * HS_ + n * 16 + l16) * S_ + sbase + r + half * 8] = (bf16)acc[n][r];
  }
}

// ---------------- causal flash attention (64-wide t blocks) ----------------
__global__ void __launch_bounds__(256)
attn_kernel(const bf16* __restrict__ Q, const bf16* __restrict__ K,
            const bf16* __restrict__ Vt, bf16* __restrict__ AO) {
  __shared__ bf16 lds[8][16 * 64];
  const int lane  = threadIdx.x & 31;
  const int l16   = lane & 15;
  const int half  = lane >> 4;
  const int klo   = half * 8;
  const int khalf = half * 16;
  bf16* pt = lds[threadIdx.x >> 5];

  const int w  = (blockIdx.x * blockDim.x + threadIdx.x) >> 5;
  const int it = w & 127;
  const int h  = (w >> 7) & 15;
  const int b  = w >> 11;
  const size_t bh = (size_t)(b * H_ + h);
  const int sbase = it * 16;

  const bf16* qrow = Q + (bh * S_ + sbase + l16) * HS_;
  v16bf aQ0 = frag_ld(qrow + klo, qrow + 16 + klo);
  v16bf aQ1 = frag_ld(qrow + 32 + klo, qrow + 48 + klo);

  v8f o[4];
  float m_run[8], l_run[8];
#pragma unroll
  for (int n = 0; n < 4; ++n) o[n] = zero8();
#pragma unroll
  for (int r = 0; r < 8; ++r) { m_run[r] = -1e30f; l_run[r] = 0.0f; }

  const int jd = (sbase + 15) >> 6;
  for (int j = 0; j <= jd; ++j) {
    const int tbase = j * 64;

    // ---- QK^T (Q pre-scaled by 1/8) ----
    v8f s[4];
#pragma unroll
    for (int stt = 0; stt < 4; ++stt) {
      const bf16* kr = K + (bh * S_ + tbase + stt * 16 + l16) * HS_;
      v16bf blo = frag_ld(kr + klo, kr + 16 + klo);
      v16bf bhi = frag_ld(kr + 32 + klo, kr + 48 + klo);
      s[stt] = wmma_bf16(aQ0, blo, zero8());
      s[stt] = wmma_bf16(aQ1, bhi, s[stt]);
    }

    // early-issue V fragments: latency hidden under the softmax VALU work
    v16bf vf[8];
#pragma unroll
    for (int n = 0; n < 4; ++n) {
      const bf16* vr = Vt + (bh * HS_ + n * 16 + l16) * S_ + tbase;
      vf[2 * n]     = frag_ld(vr + klo, vr + 16 + klo);
      vf[2 * n + 1] = frag_ld(vr + 32 + klo, vr + 48 + klo);
    }

    if (j == jd) {  // causal mask, diagonal block only
#pragma unroll
      for (int stt = 0; stt < 4; ++stt)
#pragma unroll
        for (int r = 0; r < 8; ++r) {
          const int srow = sbase + r + half * 8;
          if (tbase + stt * 16 + l16 > srow) s[stt][r] = -1e30f;
        }
    }

    // ---- online softmax over 64 columns ----
#pragma unroll
    for (int r = 0; r < 8; ++r) {
      float mx = fmaxf(fmaxf(s[0][r], s[1][r]), fmaxf(s[2][r], s[3][r]));
#pragma unroll
      for (int d = 1; d < 16; d <<= 1) mx = fmaxf(mx, __shfl_xor(mx, d, 32));
      const float mnew  = fmaxf(m_run[r], mx);
      const float alpha = __expf(m_run[r] - mnew);
      float rs = 0.0f;
      const int row = r + half * 8;
#pragma unroll
      for (int stt = 0; stt < 4; ++stt) {
        const float p = __expf(s[stt][r] - mnew);
        rs += p;
        pt[row * 64 + stt * 16 + l16] = (bf16)p;
      }
#pragma unroll
      for (int d = 1; d < 16; d <<= 1) rs += __shfl_xor(rs, d, 32);
      l_run[r] = fmaf(l_run[r], alpha, rs);
      m_run[r] = mnew;
#pragma unroll
      for (int n = 0; n < 4; ++n) o[n][r] *= alpha;
    }
    wave_lds_sync();

    const bf16* pr = pt + l16 * 64;
    v16bf aP0 = frag_ld(pr + klo, pr + 16 + klo);
    v16bf aP1 = frag_ld(pr + 32 + klo, pr + 48 + klo);

    // ---- PV accumulate ----
#pragma unroll
    for (int n = 0; n < 4; ++n) {
      o[n] = wmma_bf16(aP0, vf[2 * n], o[n]);
      o[n] = wmma_bf16(aP1, vf[2 * n + 1], o[n]);
    }
    wave_lds_sync();
  }

#pragma unroll
  for (int r = 0; r < 8; ++r) {
    const float inv = __builtin_amdgcn_rcpf(l_run[r]);
#pragma unroll
    for (int n = 0; n < 4; ++n) {
      AO[(size_t)(b * S_ + sbase + r + half * 8) * D_ + h * HS_ + n * 16 + l16] =
          (bf16)(o[n][r] * inv);
    }
  }
}

// ---------------- output projection ----------------
// Block (16 waves) = one 16-row tile of [B*S, D], TDM-staged; each wave owns a
// 64-wide column group; weights register-double-buffered.
__global__ void __launch_bounds__(512)
oproj_kernel(const bf16* __restrict__ AO, const bf16* __restrict__ wot,
             const float* __restrict__ bo, float* __restrict__ out) {
  extern __shared__ bf16 as_[];  // [16][1024], LDS offset 0
  const int lane  = threadIdx.x & 31;
  const int l16   = lane & 15;
  const int half  = lane >> 4;
  const int klo   = half * 8;
  const int khalf = half * 16;
  const int wave  = threadIdx.x >> 5;   // 0..15
  const int sbase = blockIdx.x * 16;
  const int nbase = wave * 64;

  if (wave == 0) {
    tdm_load_2d_bf16(0u, AO + (size_t)sbase * D_, D_, 16, D_, B_ * S_, D_);
    __builtin_amdgcn_s_wait_tensorcnt(0);
  }
  __syncthreads();

  v8f acc[4];
#pragma unroll
  for (int n = 0; n < 4; ++n) acc[n] = zero8();

  const bf16* arow  = as_ + l16 * D_;
  const bf16* wbase = wot + (size_t)(nbase + l16) * D_;

  v16bf bwa[4], bwb[4];
  auto ldB = [&](int kb, v16bf* dst) {
#pragma unroll
    for (int n = 0; n < 4; ++n) {
      const bf16* wr = wbase + (size_t)(n * 16) * D_ + kb;
      dst[n] = frag_ld(wr + klo, wr + 16 + klo);
    }
  };
  auto step = [&](int kb, const v16bf* bw) {
    v16bf aA = frag_ld(arow + kb + klo, arow + kb + 16 + klo);  // LDS
#pragma unroll
    for (int n = 0; n < 4; ++n) acc[n] = wmma_bf16(aA, bw[n], acc[n]);
  };

  ldB(0, bwa);
#pragma unroll 1
  for (int kk = 0; kk < D_ / 32; kk += 2) {
    ldB((kk + 1) * 32, bwb);
    step(kk * 32, bwa);
    if (kk + 2 < D_ / 32) ldB((kk + 2) * 32, bwa);
    step((kk + 1) * 32, bwb);
  }

#pragma unroll
  for (int n = 0; n < 4; ++n) {
    const float bv = bo[nbase + n * 16 + l16];
#pragma unroll
    for (int r = 0; r < 8; ++r)
      out[(size_t)(sbase + r + half * 8) * D_ + nbase + n * 16 + l16] = acc[n][r] + bv;
  }
}

// ---------------- launch ----------------
extern "C" void kernel_launch(void* const* d_in, const int* in_sizes, int n_in,
                              void* d_out, int out_size, void* d_ws, size_t ws_size,
                              hipStream_t stream) {
  const float* x  = (const float*)d_in[0];
  const float* Wq = (const float*)d_in[1];
  const float* Wk = (const float*)d_in[2];
  const float* Wv = (const float*)d_in[3];
  const float* Wo = (const float*)d_in[4];
  const float* bo = (const float*)d_in[5];
  float* out = (float*)d_out;

  char* ws = (char*)d_ws;
  const size_t MB = 1024 * 1024;
  bf16* xb  = (bf16*)(ws);             // 16 MB : x bf16 [B][S][D]
  bf16* wt  = (bf16*)(ws + 16 * MB);   //  6 MB : Wq/k/v^T bf16 [3][H][HS][D]
  bf16* wot = (bf16*)(ws + 22 * MB);   //  2 MB : Wo^T bf16 [D][D]
  bf16* Qb  = (bf16*)(ws + 24 * MB);   // 16 MB : Q bf16 (pre-scaled by 1/8)
  bf16* Kb  = (bf16*)(ws + 40 * MB);   // 16 MB : K bf16
  bf16* Vt  = (bf16*)(ws + 56 * MB);   // 16 MB : V^T bf16 [B][H][HS][S]
  bf16* AO  = (bf16*)(ws + 72 * MB);   // 16 MB : attn out bf16 [B][S][D]

  cvt_x_kernel<<<(B_ * S_ * D_ / 4 + 255) / 256, 256, 0, stream>>>(x, xb, B_ * S_ * D_ / 4);
  cvt_w_kernel<<<(3 * H_ * HS_ * D_ + 255) / 256, 256, 0, stream>>>(Wq, Wk, Wv, wt);
  cvt_wo_kernel<<<(D_ * D_ + 255) / 256, 256, 0, stream>>>(Wo, wot);

  qkv_kernel<<<2048, 384, 16 * D_ * sizeof(bf16), stream>>>(xb, wt, Qb, Kb, Vt);
  attn_kernel<<<1024, 256, 0, stream>>>(Qb, Kb, Vt, AO);
  oproj_kernel<<<512, 512, 16 * D_ * sizeof(bf16), stream>>>(AO, wot, bo, out);
}
